// Qwen3MoeModel_24833500906105
// MI455X (gfx1250) — matrix operand order repaired and checked by reading, not compiled, not run on hardware
//
#include <hip/hip_runtime.h>
#include <cstdint>
#include <cstddef>

// ---------------------------------------------------------------------------
// Qwen3-MoE layer for MI455X (gfx1250, wave32, WMMA, async-to-LDS).
// T=4096, H=2048, E=64, F=768, K=8, capacity C=1024.
// Grouped GEMMs (~0.62 TFLOP) on v_wmma_f32_16x16x32_bf16, f32 accum.
// B weights pre-swizzled into WMMA fragment layout (L2-resident re-reads);
// A tiles streamed via global_load_async_to_lds_b128, double-buffered.
// 512-thread blocks / 2x8 wave tiling keep VGPRs < 256 (no spills, no MSB).
// ---------------------------------------------------------------------------

#define T_TOK 4096
#define H_DIM 2048
#define E_EXP 64
#define F_DIM 768
#define TOPK  8
#define CAP   1024

#define LDA 40  // LDS A row stride (bf16): 32 data + 8 pad -> 80B rows, 16B aligned

typedef __attribute__((ext_vector_type(16))) __bf16 bf16x16;
typedef __attribute__((ext_vector_type(8)))  float  f32x8;

union Frag16 {
    uint4   u[2];
    bf16x16 v;
    __bf16  h[16];
};

// ---- CDNA5 async global->LDS (ASYNCcnt) -----------------------------------
__device__ __forceinline__ void async_load_b128(const void* gaddr, unsigned lds_off) {
    asm volatile("global_load_async_to_lds_b128 %0, %1, off"
                 :: "v"(lds_off), "v"(gaddr) : "memory");
}
__device__ __forceinline__ void wait_async0() {
    asm volatile("s_wait_asynccnt 0x0" ::: "memory");
}
__device__ __forceinline__ unsigned lds_off_of(const void* p) {
    return (unsigned)(uintptr_t)p;  // flat LDS aperture: addr[31:0] == LDS offset
}

// ---- fp32 -> bf16 (x activations, n multiple of 4) ------------------------
__global__ void cvt_bf16_kernel(const float* __restrict__ src,
                                __bf16* __restrict__ dst, long long n) {
    long long i = (long long)blockIdx.x * blockDim.x + threadIdx.x;
    long long stride = (long long)gridDim.x * blockDim.x;
    for (long long j = i * 4; j < n; j += stride * 4) {
        float4 v = *(const float4*)(src + j);
        dst[j + 0] = (__bf16)v.x;
        dst[j + 1] = (__bf16)v.y;
        dst[j + 2] = (__bf16)v.z;
        dst[j + 3] = (__bf16)v.w;
    }
}

__global__ void zero_kernel(float* __restrict__ p, long long n) {
    long long i = (long long)blockIdx.x * blockDim.x + threadIdx.x;
    long long stride = (long long)gridDim.x * blockDim.x;
    for (long long j = i; j < n; j += stride) p[j] = 0.0f;
}

// ---- weight swizzle: fp32 [E][K][N] -> bf16 WMMA B-fragment layout --------
// dst fragment (e, nblk=n/16, kblk=k/32): 512 bf16; lane ln holds
// n = nblk*16 + (ln&15), k = kblk*32 + (ln>>4)*16 + i, i=0..15 (32B contiguous).
__global__ __launch_bounds__(256)
void swizzle_b_kernel(const float* __restrict__ src, __bf16* __restrict__ dst,
                      int K, int N) {
    __shared__ float tile[32][132];
    const int e = blockIdx.z, kblk = blockIdx.y, nc = blockIdx.x;  // nc: 128-col chunk
    const float* s = src + (size_t)e * K * N + (size_t)kblk * 32 * N + nc * 128;
    const int tid = threadIdx.x;
    for (int i = tid; i < 1024; i += 256) {          // 32 rows x 32 float4
        int k = i >> 5, n4 = i & 31;
        float4 v = *(const float4*)(s + (size_t)k * N + n4 * 4);
        tile[k][n4 * 4 + 0] = v.x; tile[k][n4 * 4 + 1] = v.y;
        tile[k][n4 * 4 + 2] = v.z; tile[k][n4 * 4 + 3] = v.w;
    }
    __syncthreads();
    const int nb = tid >> 5, lane = tid & 31;        // 8 fragments per block
    const int n = nb * 16 + (lane & 15);
    const int k0 = (lane >> 4) * 16;
    Frag16 f;
    for (int i = 0; i < 16; ++i) f.h[i] = (__bf16)tile[k0 + i][n];
    size_t nblk_g = (size_t)e * (N / 16) + nc * 8 + nb;
    uint4* d = (uint4*)(dst + (nblk_g * (size_t)(K / 32) + kblk) * 512 + lane * 16);
    d[0] = f.u[0];
    d[1] = f.u[1];
}

// ---- router: logits -> softmax -> top-8 -> renormalize --------------------
__global__ __launch_bounds__(256)
void router_kernel(const float* __restrict__ x, const float* __restrict__ gate_w,
                   int* __restrict__ topi, float* __restrict__ topw) {
    __shared__ float xs[H_DIM];
    __shared__ float partial[E_EXP][4];
    __shared__ float logits[E_EXP];
    const int t = blockIdx.x;
    const int tid = threadIdx.x;

    const float4* xrow = (const float4*)(x + (size_t)t * H_DIM);
    for (int i = tid; i < H_DIM / 4; i += 256) ((float4*)xs)[i] = xrow[i];
    __syncthreads();

    const int e = tid >> 2, part = tid & 3;
    const float4* gw = (const float4*)(gate_w + (size_t)e * H_DIM + part * 512);
    const float4* xp = (const float4*)(xs + part * 512);
    float acc = 0.0f;
    for (int i = 0; i < 128; ++i) {
        float4 a = xp[i], b = gw[i];
        acc += a.x * b.x + a.y * b.y + a.z * b.z + a.w * b.w;
    }
    partial[e][part] = acc;
    __syncthreads();
    if (tid < E_EXP)
        logits[tid] = partial[tid][0] + partial[tid][1] + partial[tid][2] + partial[tid][3];
    __syncthreads();

    if (tid == 0) {
        float mx = -1e30f;
        for (int i = 0; i < E_EXP; ++i) mx = fmaxf(mx, logits[i]);
        float p[E_EXP];
        float sum = 0.0f;
        for (int i = 0; i < E_EXP; ++i) { p[i] = __expf(logits[i] - mx); sum += p[i]; }
        float inv = 1.0f / sum;
        int idx[TOPK]; float val[TOPK]; float wsum = 0.0f;
        for (int kk = 0; kk < TOPK; ++kk) {
            int bi = 0; float bv = -1.0f;
            for (int i = 0; i < E_EXP; ++i)
                if (p[i] > bv) { bv = p[i]; bi = i; }
            idx[kk] = bi; val[kk] = bv * inv; wsum += bv * inv; p[bi] = -2.0f;
        }
        float winv = 1.0f / wsum;
        for (int kk = 0; kk < TOPK; ++kk) {
            topi[t * TOPK + kk] = idx[kk];
            topw[t * TOPK + kk] = val[kk] * winv;
        }
    }
}

// ---- dispatch: deterministic token-order capacity assignment --------------
__global__ void dispatch_kernel(const int* __restrict__ topi,
                                int* __restrict__ token_flat) {
    const int e = blockIdx.x;
    const int lane = threadIdx.x;            // one wave
    const unsigned lt = (1u << lane) - 1u;
    int base = 0;
    for (int f0 = 0; f0 < T_TOK * TOPK; f0 += 32) {
        int flat = f0 + lane;
        bool m = (topi[flat] == e);
        unsigned mask = __builtin_amdgcn_ballot_w32(m);
        if (m) {
            int pos = base + __builtin_popcount(mask & lt);
            if (pos < CAP) token_flat[e * CAP + pos] = flat;
        }
        base += __builtin_popcount(mask);
    }
    for (int i = base + lane; i < CAP; i += 32) token_flat[e * CAP + i] = -1;
}

// ---- A fragment from padded LDS tile (contiguous k = half*16 + i, same order as B swizzle) ---------
__device__ __forceinline__ bf16x16 load_frag_a(const __bf16* lds, int row_base, int lane) {
    const int m = lane & 15, half = lane >> 4;
    const __bf16* p = lds + (row_base + m) * LDA + half * 16;
    Frag16 f;
    f.u[0] = ((const uint4*)p)[0];       // K = half*16 .. +7
    f.u[1] = *(const uint4*)(p + 8);     // K = half*16+8 .. +7
    return f.v;
}
// B fragment direct from swizzled global (32B per lane)
__device__ __forceinline__ bf16x16 load_frag_b_g(const __bf16* p) {
    Frag16 f;
    f.u[0] = ((const uint4*)p)[0];
    f.u[1] = ((const uint4*)p)[1];
    return f.v;
}

#define WMMA_BF16(A, B, C) \
    __builtin_amdgcn_wmma_f32_16x16x32_bf16(false, (A), false, (B), (short)0, (C), false, false)

// ---- GEMM1: gathered x @ w_gate_up (gate+up fused) -> SiLU(g)*u -> act ----
// 512 threads = 16 waves as 2(wm) x 8(wn); wave tile M64 x N16 (gate+up).
__global__ __launch_bounds__(512)
void gemm1_kernel(const __bf16* __restrict__ x_bf,
                  const __bf16* __restrict__ bsw,      // swizzled [e*96+nblk][64][512]
                  const int* __restrict__ token_flat,
                  __bf16* __restrict__ act_bf) {
    const int e = blockIdx.z, mt = blockIdx.y, ft = blockIdx.x;  // 64 x 8 x 6
    __shared__ __bf16 ldsA[2][128 * LDA];

    const int tid = threadIdx.x, lane = tid & 31, wave = tid >> 5;
    const int wm = wave >> 3, wn = wave & 7;

    // A loader mapping: thread -> (row ar, 8-elem quarter aq); one b128 each
    const int ar = tid >> 2, aq = tid & 3;
    const int aflat = token_flat[e * CAP + mt * 128 + ar];
    const __bf16* arow = (aflat >= 0)
        ? (x_bf + (size_t)(aflat >> 3) * H_DIM + aq * 8) : (const __bf16*)nullptr;
    const unsigned loff0 = lds_off_of(&ldsA[0][ar * LDA + aq * 8]);
    const unsigned loff1 = lds_off_of(&ldsA[1][ar * LDA + aq * 8]);

    // zero gathered-invalid rows once (never async-written afterwards)
    if (!arow) {
        uint4 z = make_uint4(0, 0, 0, 0);
        *(uint4*)(&ldsA[0][ar * LDA + aq * 8]) = z;
        *(uint4*)(&ldsA[1][ar * LDA + aq * 8]) = z;
    }

    // B fragment bases (gate: nblk, up: nblk+48), KB=64 kblks, 512 elems apart
    const int KB = H_DIM / 32;   // 64
    const size_t nblk0 = (size_t)e * 96 + ft * 8 + wn;
    const __bf16* bg = bsw + nblk0 * (size_t)KB * 512 + lane * 16;
    const __bf16* bu = bg + (size_t)48 * KB * 512;

    f32x8 accg[4] = {};
    f32x8 accu[4] = {};

    if (arow) async_load_b128(arow, loff0);   // prologue: stream first A tile

    for (int kb = 0; kb < KB; ++kb) {
        wait_async0();
        __syncthreads();
        if (kb + 1 < KB && arow)              // stream next tile into other buffer
            async_load_b128(arow + (kb + 1) * 32, (kb & 1) ? loff0 : loff1);

        const __bf16* bufA = ldsA[kb & 1];
        const int koff = kb * 512;

        bf16x16 a0 = load_frag_a(bufA, wm * 64 + 0,  lane);
        bf16x16 a1 = load_frag_a(bufA, wm * 64 + 16, lane);
        bf16x16 a2 = load_frag_a(bufA, wm * 64 + 32, lane);
        bf16x16 a3 = load_frag_a(bufA, wm * 64 + 48, lane);
        bf16x16 g  = load_frag_b_g(bg + koff);
        bf16x16 u  = load_frag_b_g(bu + koff);

        accg[0] = WMMA_BF16(a0, g, accg[0]);
        accg[1] = WMMA_BF16(a1, g, accg[1]);
        accg[2] = WMMA_BF16(a2, g, accg[2]);
        accg[3] = WMMA_BF16(a3, g, accg[3]);
        accu[0] = WMMA_BF16(a0, u, accu[0]);
        accu[1] = WMMA_BF16(a1, u, accu[1]);
        accu[2] = WMMA_BF16(a2, u, accu[2]);
        accu[3] = WMMA_BF16(a3, u, accu[3]);
    }

    // epilogue: act[e][slot][f] = silu(gate) * up
    const int row0 = mt * 128 + wm * 64 + ((lane >> 4) * 8);
    const int col  = ft * 128 + wn * 16 + (lane & 15);
    for (int fm = 0; fm < 4; ++fm) {
        for (int r = 0; r < 8; ++r) {
            int row = row0 + fm * 16 + r;
            float g = accg[fm][r];
            float u = accu[fm][r];
            float sg = g / (1.0f + __expf(-g));
            act_bf[((size_t)e * CAP + row) * F_DIM + col] = (__bf16)(sg * u);
        }
    }
}

// ---- GEMM2: act @ w_down, weighted scatter-add into out -------------------
// 512 threads = 16 waves as 2(wm) x 8(wn); wave tile M64 x N16.
__global__ __launch_bounds__(512)
void gemm2_kernel(const __bf16* __restrict__ act_bf,
                  const __bf16* __restrict__ bsw,      // swizzled [e*128+nblk][24][512]
                  const int* __restrict__ token_flat,
                  const float* __restrict__ topw,
                  float* __restrict__ out) {
    const int e = blockIdx.z, mt = blockIdx.y, nt = blockIdx.x;  // 64 x 8 x 16
    __shared__ __bf16 ldsA[2][128 * LDA];

    const int tid = threadIdx.x, lane = tid & 31, wave = tid >> 5;
    const int wm = wave >> 3, wn = wave & 7;

    const int ar = tid >> 2, aq = tid & 3;
    const __bf16* arow = act_bf + ((size_t)e * CAP + mt * 128 + ar) * F_DIM + aq * 8;
    const unsigned loff0 = lds_off_of(&ldsA[0][ar * LDA + aq * 8]);
    const unsigned loff1 = lds_off_of(&ldsA[1][ar * LDA + aq * 8]);

    const int KB = F_DIM / 32;   // 24
    const size_t nblk0 = (size_t)e * 128 + nt * 8 + wn;
    const __bf16* bp = bsw + nblk0 * (size_t)KB * 512 + lane * 16;

    f32x8 acc[4] = {};

    async_load_b128(arow, loff0);

    for (int kb = 0; kb < KB; ++kb) {
        wait_async0();
        __syncthreads();
        if (kb + 1 < KB)
            async_load_b128(arow + (kb + 1) * 32, (kb & 1) ? loff0 : loff1);

        const __bf16* bufA = ldsA[kb & 1];
        const int koff = kb * 512;

        bf16x16 a0 = load_frag_a(bufA, wm * 64 + 0,  lane);
        bf16x16 a1 = load_frag_a(bufA, wm * 64 + 16, lane);
        bf16x16 a2 = load_frag_a(bufA, wm * 64 + 32, lane);
        bf16x16 a3 = load_frag_a(bufA, wm * 64 + 48, lane);
        bf16x16 b  = load_frag_b_g(bp + koff);

        acc[0] = WMMA_BF16(a0, b, acc[0]);
        acc[1] = WMMA_BF16(a1, b, acc[1]);
        acc[2] = WMMA_BF16(a2, b, acc[2]);
        acc[3] = WMMA_BF16(a3, b, acc[3]);
    }

    const int slot0 = mt * 128 + wm * 64 + ((lane >> 4) * 8);
    const int ncol  = nt * 128 + wn * 16 + (lane & 15);
    for (int fm = 0; fm < 4; ++fm) {
        for (int r = 0; r < 8; ++r) {
            int slot = slot0 + fm * 16 + r;
            int flat = token_flat[e * CAP + slot];
            if (flat < 0) continue;
            float w = topw[flat];
            atomicAdd(out + (size_t)(flat >> 3) * H_DIM + ncol, acc[fm][r] * w);
        }
    }
}

// ---------------------------------------------------------------------------
extern "C" void kernel_launch(void* const* d_in, const int* in_sizes, int n_in,
                              void* d_out, int out_size, void* d_ws, size_t ws_size,
                              hipStream_t stream) {
    const float* x      = (const float*)d_in[0];   // [T, H]
    const float* gate_w = (const float*)d_in[1];   // [E, H]
    const float* w_gu   = (const float*)d_in[2];   // [E, H, 2F]
    const float* w_dn   = (const float*)d_in[3];   // [E, F, H]
    float* out = (float*)d_out;                    // [T, H]

    char* ws = (char*)d_ws;
    size_t off = 0;
    auto salloc = [&](size_t bytes) -> void* {
        off = (off + 255) & ~(size_t)255;
        void* p = ws + off;
        off += bytes;
        return p;
    };
    __bf16* x_bf    = (__bf16*)salloc((size_t)T_TOK * H_DIM * 2);
    __bf16* bswg    = (__bf16*)salloc((size_t)E_EXP * H_DIM * 2 * F_DIM * 2);
    __bf16* bswd    = (__bf16*)salloc((size_t)E_EXP * F_DIM * H_DIM * 2);
    __bf16* act_bf  = (__bf16*)salloc((size_t)E_EXP * CAP * F_DIM * 2);
    int*    topi    = (int*)   salloc((size_t)T_TOK * TOPK * 4);
    float*  topw    = (float*) salloc((size_t)T_TOK * TOPK * 4);
    int*    tokflat = (int*)   salloc((size_t)E_EXP * CAP * 4);

    cvt_bf16_kernel<<<2048, 256, 0, stream>>>(x, x_bf, (long long)T_TOK * H_DIM);
    swizzle_b_kernel<<<dim3(12, 64, E_EXP), 256, 0, stream>>>(w_gu, bswg, H_DIM, 2 * F_DIM);
    swizzle_b_kernel<<<dim3(16, 24, E_EXP), 256, 0, stream>>>(w_dn, bswd, F_DIM, H_DIM);
    zero_kernel<<<2048, 256, 0, stream>>>(out, (long long)T_TOK * H_DIM);

    router_kernel<<<T_TOK, 256, 0, stream>>>(x, gate_w, topi, topw);
    dispatch_kernel<<<E_EXP, 32, 0, stream>>>(topi, tokflat);

    gemm1_kernel<<<dim3(6, 8, E_EXP), 512, 0, stream>>>(x_bf, bswg, tokflat, act_bf);
    gemm2_kernel<<<dim3(16, 8, E_EXP), 512, 0, stream>>>(act_bf, bswd, tokflat, topw, out);
}
